// MLAOnlyBlock_65412351918131
// MI455X (gfx1250) — compile-verified, matrix operand order checked by reading, not run on hardware
//
#include <hip/hip_runtime.h>
#include <hip/hip_bf16.h>
#include <math.h>

// ---------------- types ----------------
typedef __attribute__((ext_vector_type(16))) __bf16 v16bf;
typedef __attribute__((ext_vector_type(8)))  float  v8f;

union FragU { v16bf v; uint4 q[2]; };

// Load a 16-lane-wide WMMA fragment (A or B^T style) from LDS.
// Per ISA 7.12.2 (16-bit A 16x32): lanes 0-15 hold K=[0..7],[16..23];
// lanes 16-31 hold K=[8..15],[24..31]  -> two b128 LDS reads per lane.
__device__ inline v16bf load_frag(const __bf16* base, int stride, int idx_base, int k_base) {
    int lane = threadIdx.x & 31;
    const __bf16* p = base + (size_t)(idx_base + (lane & 15)) * stride
                           + k_base + ((lane >> 4) & 1) * 8;
    FragU f;
    f.q[0] = *(const uint4*)p;
    f.q[1] = *(const uint4*)(p + 16);
    return f.v;
}

// ---------------- CDNA5 async global->LDS copy (ASYNCcnt path) ----------------
// GLOBAL_LOAD_ASYNC_TO_LDS_B128: per-lane 16B memory -> LDS, no VGPR round trip.
// LDS address = low 32 bits of flat shared pointer.
__device__ inline void async_ld_b128(void* lds_ptr, const void* gaddr) {
    uint32_t l = (uint32_t)(size_t)lds_ptr;
    asm volatile("global_load_async_to_lds_b128 %0, %1, off"
                 :: "v"(l), "v"(gaddr) : "memory");
}
__device__ inline void wait_async0() {
    asm volatile("s_wait_asynccnt 0x0" ::: "memory");
}
__device__ inline void wait_async4() {   // allow 4 newest in flight (next tile)
    asm volatile("s_wait_asynccnt 0x4" ::: "memory");
}

// ---------------- weight transpose + fp32 -> bf16 ----------------
// W [K][N] f32 (row-major)  ->  Wt [N][K] bf16 (row-major)
__global__ void transpose_cvt_kernel(const float* __restrict__ W, __bf16* __restrict__ Wt,
                                     int K, int N) {
    __shared__ float t[32][33];
    int k0 = blockIdx.y * 32, n0 = blockIdx.x * 32;
    int tx = threadIdx.x, ty = threadIdx.y; // block 32x8
    for (int i = 0; i < 32; i += 8)
        t[ty + i][tx] = W[(size_t)(k0 + ty + i) * N + (n0 + tx)];
    __syncthreads();
    for (int i = 0; i < 32; i += 8)
        Wt[(size_t)(n0 + ty + i) * K + (k0 + tx)] = (__bf16)t[tx][ty + i];
}

// ---------------- RMSNorm (f32 in -> bf16 out) ----------------
__global__ __launch_bounds__(256)
void rmsnorm_bf16_kernel(const float* __restrict__ x, const float* __restrict__ w,
                         __bf16* __restrict__ out, int n, int in_ld, int out_ld) {
    int row = blockIdx.x;
    const float* xr = x + (size_t)row * in_ld;
    float ss = 0.f;
    for (int j = threadIdx.x; j < n; j += 256) { float v = xr[j]; ss += v * v; }
    for (int off = 16; off; off >>= 1) ss += __shfl_xor(ss, off);
    __shared__ float red[8];
    if ((threadIdx.x & 31) == 0) red[threadIdx.x >> 5] = ss;
    __syncthreads();
    if (threadIdx.x == 0) { float t = 0.f; for (int i = 0; i < 8; i++) t += red[i]; red[0] = t; }
    __syncthreads();
    float r = rsqrtf(red[0] / (float)n + 1e-5f);
    for (int j = threadIdx.x; j < n; j += 256)
        out[(size_t)row * out_ld + j] = (__bf16)(xr[j] * r * w[j]);
}

// ---------------- WMMA GEMM: C[M,N] = A[M,K] x Bt[N,K]^T ----------------
// Double-buffered LDS, async global->LDS staging.
// EPI: 0 = f32 store, 1 = bf16 store, 2 = bias+GELU -> bf16,
//      3 = +res -> f32, 4 = +bias+res -> f32
template<int EPI>
__global__ __launch_bounds__(256)
void gemm_bf16_kernel(const __bf16* __restrict__ A, const __bf16* __restrict__ Bt,
                      float* __restrict__ outF, __bf16* __restrict__ outB,
                      const float* __restrict__ bias, const float* __restrict__ res,
                      int M, int N, int K) {
    __shared__ __align__(16) __bf16 As[2][128 * 40];
    __shared__ __align__(16) __bf16 Bs[2][128 * 40];
    int tid = threadIdx.x, lane = tid & 31, wid = tid >> 5;
    int m0 = blockIdx.y * 128, n0 = blockIdx.x * 128;
    int wm = wid & 3, wn = wid >> 2;           // 4x2 wave grid: 32 rows x 64 cols per wave

    // per-thread staging coordinates (2 x uint4 per tile per array)
    int r0 = tid >> 2, kv0 = tid & 3;
    int r1 = (tid + 256) >> 2, kv1 = (tid + 256) & 3;

    v8f acc[2][4];
    for (int a = 0; a < 2; a++) for (int b = 0; b < 4; b++) acc[a][b] = {};

    int nk = K >> 5;
    // prologue: stage tile 0 into buffer 0
    {
        async_ld_b128(&As[0][r0 * 40 + kv0 * 8], &A[(size_t)(m0 + r0) * K + kv0 * 8]);
        async_ld_b128(&As[0][r1 * 40 + kv1 * 8], &A[(size_t)(m0 + r1) * K + kv1 * 8]);
        async_ld_b128(&Bs[0][r0 * 40 + kv0 * 8], &Bt[(size_t)(n0 + r0) * K + kv0 * 8]);
        async_ld_b128(&Bs[0][r1 * 40 + kv1 * 8], &Bt[(size_t)(n0 + r1) * K + kv1 * 8]);
    }

    for (int ki = 0; ki < nk; ki++) {
        int buf = ki & 1;
        if (ki + 1 < nk) {
            int kt = (ki + 1) << 5, nb = buf ^ 1;
            async_ld_b128(&As[nb][r0 * 40 + kv0 * 8], &A[(size_t)(m0 + r0) * K + kt + kv0 * 8]);
            async_ld_b128(&As[nb][r1 * 40 + kv1 * 8], &A[(size_t)(m0 + r1) * K + kt + kv1 * 8]);
            async_ld_b128(&Bs[nb][r0 * 40 + kv0 * 8], &Bt[(size_t)(n0 + r0) * K + kt + kv0 * 8]);
            async_ld_b128(&Bs[nb][r1 * 40 + kv1 * 8], &Bt[(size_t)(n0 + r1) * K + kt + kv1 * 8]);
            wait_async4();   // previous tile complete; next tile may stay in flight
        } else {
            wait_async0();
        }
        __syncthreads();

        v16bf bf[4];
        for (int tn = 0; tn < 4; tn++) bf[tn] = load_frag(Bs[buf], 40, wn * 64 + tn * 16, 0);
        for (int tm = 0; tm < 2; tm++) {
            v16bf af = load_frag(As[buf], 40, wm * 32 + tm * 16, 0);
            for (int tn = 0; tn < 4; tn++)
                acc[tm][tn] = __builtin_amdgcn_wmma_f32_16x16x32_bf16(
                    false, af, false, bf[tn], (short)0, acc[tm][tn], false, false);
        }
        __syncthreads();
    }

    int half = (lane >> 4) & 1, cl = lane & 15;
    for (int tm = 0; tm < 2; tm++)
        for (int tn = 0; tn < 4; tn++)
            for (int i = 0; i < 8; i++) {
                int r = m0 + wm * 32 + tm * 16 + i + half * 8;
                int c = n0 + wn * 64 + tn * 16 + cl;
                if (c >= N) continue;
                float v = acc[tm][tn][i];
                size_t idx = (size_t)r * N + c;
                if (EPI == 0) outF[idx] = v;
                else if (EPI == 1) outB[idx] = (__bf16)v;
                else if (EPI == 2) {
                    v += bias[c];
                    v = 0.5f * v * (1.0f + erff(v * 0.70710678118654752f));
                    outB[idx] = (__bf16)v;
                } else if (EPI == 3) outF[idx] = res[idx] + v;
                else outF[idx] = res[idx] + v + bias[c];
            }
}

// ---------------- RoPE pack kernels ----------------
// Reference rope: out[j]    = x[2j]*cos_j   - x[2j+1]*sin_j
//                 out[32+j] = x[2j+1]*cos_j + x[2j]*sin_j      (j < 32, d=64)
__global__ void pack_q_kernel(const float* __restrict__ q, __bf16* __restrict__ Qf) {
    int s = blockIdx.x, h = blockIdx.y, t = threadIdx.x;
    const float* qr = q + (size_t)s * 3072 + h * 192;
    __bf16* o = Qf + ((size_t)h * 2048 + s) * 192;
    if (t < 128) {
        o[t] = (__bf16)qr[t];
    } else if (t < 160) {
        int p = t - 128;
        float inv = __powf(10000.0f, -(float)p / 32.0f);
        float f = (float)s * inv, c = __cosf(f), sn = __sinf(f);
        float x0 = qr[128 + 2 * p], x1 = qr[128 + 2 * p + 1];
        o[128 + p] = (__bf16)(x0 * c - x1 * sn);
        o[160 + p] = (__bf16)(x1 * c + x0 * sn);
    }
}

__global__ void pack_k_kernel(const float* __restrict__ kv, const float* __restrict__ kv_a,
                              __bf16* __restrict__ Kf) {
    int s = blockIdx.x, h = blockIdx.y, t = threadIdx.x;
    __bf16* o = Kf + ((size_t)h * 2048 + s) * 192;
    if (t < 128) {
        o[t] = (__bf16)kv[(size_t)s * 4096 + h * 256 + t];
    } else if (t < 160) {
        int p = t - 128;
        float inv = __powf(10000.0f, -(float)p / 32.0f);
        float f = (float)s * inv, c = __cosf(f), sn = __sinf(f);
        float x0 = kv_a[(size_t)s * 576 + 512 + 2 * p];
        float x1 = kv_a[(size_t)s * 576 + 512 + 2 * p + 1];
        o[128 + p] = (__bf16)(x0 * c - x1 * sn);
        o[160 + p] = (__bf16)(x1 * c + x0 * sn);
    }
}

// V transposed per head: Vt[h][d][s]
__global__ void pack_v_kernel(const float* __restrict__ kv, __bf16* __restrict__ Vt) {
    int s = blockIdx.x, h = blockIdx.y, d = threadIdx.x; // 128 threads
    Vt[((size_t)h * 128 + d) * 2048 + s] = (__bf16)kv[(size_t)s * 4096 + h * 256 + 128 + d];
}

// ---------------- Flash attention (causal, per head) ----------------
// grid (S/64, H), block 128 (4 waves); wave w owns 16 query rows.
__global__ __launch_bounds__(128)
void attn_kernel(const __bf16* __restrict__ Qf, const __bf16* __restrict__ Kf,
                 const __bf16* __restrict__ Vt, __bf16* __restrict__ Out) {
    int h = blockIdx.y, qblk = blockIdx.x;
    int tid = threadIdx.x, lane = tid & 31, wid = tid >> 5;
    int qbase = qblk * 64;
    int qrow0 = qbase + wid * 16;

    __shared__ __align__(16) __bf16 Qs[4][16 * 200];
    __shared__ __align__(16) __bf16 Ks[32 * 200];
    __shared__ __align__(16) __bf16 Vs[128 * 40];
    __shared__ __align__(16) __bf16 Ps[4][16 * 40];

    // stage Q tile (per wave 16x192) via async copy
    const __bf16* qg = Qf + ((size_t)h * 2048 + qrow0) * 192;
    for (int i = 0; i < 12; i++) {
        int v = lane + i * 32; int r = v / 24, kv = v % 24;
        async_ld_b128(&Qs[wid][r * 200 + kv * 8], &qg[r * 192 + kv * 8]);
    }

    v8f o[8];
    for (int d = 0; d < 8; d++) o[d] = {};
    float mrow[8], lrow[8];
    for (int i = 0; i < 8; i++) { mrow[i] = -3e38f; lrow[i] = 0.f; }

    const float scale = 0.072168783648703216f; // 1/sqrt(192)
    int half = (lane >> 4) & 1, cl = lane & 15;
    int nkt = 2 * qblk + 2;

    for (int kt = 0; kt < nkt; kt++) {
        int k0 = kt * 32;
        // stage K tile 32x192 (async)
        const __bf16* kg = Kf + ((size_t)h * 2048 + k0) * 192;
        for (int i = 0; i < 6; i++) {
            int v = tid + i * 128; int r = v / 24, kv = v % 24;
            async_ld_b128(&Ks[r * 200 + kv * 8], &kg[r * 192 + kv * 8]);
        }
        // stage Vt tile 128x32 (async)
        const __bf16* vg = Vt + (size_t)h * 128 * 2048 + k0;
        for (int i = 0; i < 4; i++) {
            int v = tid + i * 128; int d = v >> 2, kv = v & 3;
            async_ld_b128(&Vs[d * 40 + kv * 8], &vg[(size_t)d * 2048 + kv * 8]);
        }
        wait_async0();
        __syncthreads();

        // scores 16x32: 2 col-tiles, K=192 -> 6 wmma steps each
        v8f s0 = {}, s1 = {};
        for (int ks = 0; ks < 6; ks++) {
            v16bf a  = load_frag(Qs[wid], 200, 0, ks * 32);
            v16bf b0 = load_frag(Ks, 200, 0, ks * 32);
            v16bf b1 = load_frag(Ks, 200, 16, ks * 32);
            s0 = __builtin_amdgcn_wmma_f32_16x16x32_bf16(false, a, false, b0, (short)0, s0, false, false);
            s1 = __builtin_amdgcn_wmma_f32_16x16x32_bf16(false, a, false, b1, (short)0, s1, false, false);
        }

        // online softmax over 32 keys
        float alpha[8];
        for (int i = 0; i < 8; i++) {
            int r = qrow0 + i + half * 8;
            int c0 = k0 + cl, c1 = k0 + 16 + cl;
            float a0 = (c0 > r) ? -3e38f : s0[i] * scale;
            float a1 = (c1 > r) ? -3e38f : s1[i] * scale;
            float t = fmaxf(a0, a1);
            for (int off = 1; off <= 8; off <<= 1) t = fmaxf(t, __shfl_xor(t, off));
            float mn = fmaxf(mrow[i], t);
            float p0 = __expf(a0 - mn), p1 = __expf(a1 - mn);
            float sum = p0 + p1;
            for (int off = 1; off <= 8; off <<= 1) sum += __shfl_xor(sum, off);
            alpha[i] = __expf(mrow[i] - mn);
            lrow[i] = lrow[i] * alpha[i] + sum;
            mrow[i] = mn;
            // probs to LDS (C-layout -> A-fragment layout round trip; same-wave DS RAW is in-order)
            int rr = i + half * 8;
            Ps[wid][rr * 40 + cl]      = (__bf16)p0;
            Ps[wid][rr * 40 + 16 + cl] = (__bf16)p1;
        }
        for (int d = 0; d < 8; d++)
            for (int i = 0; i < 8; i++) o[d][i] *= alpha[i];

        // O += P(16x32) x V(32x128)
        v16bf pa = load_frag(Ps[wid], 40, 0, 0);
        for (int dt = 0; dt < 8; dt++) {
            v16bf vb = load_frag(Vs, 40, dt * 16, 0);
            o[dt] = __builtin_amdgcn_wmma_f32_16x16x32_bf16(false, pa, false, vb, (short)0, o[dt], false, false);
        }
        __syncthreads();
    }

    // finalize: divide by l, write bf16 out at [s][h*128+d]
    for (int dt = 0; dt < 8; dt++)
        for (int i = 0; i < 8; i++) {
            int r = qrow0 + i + half * 8;
            int c = h * 128 + dt * 16 + cl;
            Out[(size_t)r * 2048 + c] = (__bf16)(o[dt][i] / lrow[i]);
        }
}

// ---------------- orchestration ----------------
extern "C" void kernel_launch(void* const* d_in, const int* in_sizes, int n_in,
                              void* d_out, int out_size, void* d_ws, size_t ws_size,
                              hipStream_t stream) {
    (void)in_sizes; (void)n_in; (void)out_size; (void)ws_size;
    const int S = 2048, D = 2048, H = 16;
    const int QR = 1536, KVR = 512, ROPE = 64, QKD = 192, VD = 128, FF = 8192;

    const float* x        = (const float*)d_in[0];
    const float* norm1_w  = (const float*)d_in[1];
    const float* q_a_w    = (const float*)d_in[2];
    const float* q_a_ln_w = (const float*)d_in[3];
    const float* q_b_w    = (const float*)d_in[4];
    const float* kv_a_w   = (const float*)d_in[5];
    const float* kv_a_ln_w= (const float*)d_in[6];
    const float* kv_b_w   = (const float*)d_in[7];
    const float* o_w      = (const float*)d_in[8];
    const float* norm2_w  = (const float*)d_in[9];
    const float* ffn_w1   = (const float*)d_in[10];
    const float* ffn_b1   = (const float*)d_in[11];
    const float* ffn_w2   = (const float*)d_in[12];
    const float* ffn_b2   = (const float*)d_in[13];
    float* out = (float*)d_out;

    char* ws = (char*)d_ws;
    size_t off = 0;
    auto alloc = [&](size_t bytes) -> void* {
        void* p = ws + off; off += (bytes + 255) & ~(size_t)255; return p;
    };

    // bf16 transposed weights [N][K]
    __bf16* q_aT  = (__bf16*)alloc((size_t)QR * D * 2);
    __bf16* q_bT  = (__bf16*)alloc((size_t)(H * QKD) * QR * 2);
    __bf16* kv_aT = (__bf16*)alloc((size_t)(KVR + ROPE) * D * 2);
    __bf16* kv_bT = (__bf16*)alloc((size_t)(H * (128 + VD)) * KVR * 2);
    __bf16* oT    = (__bf16*)alloc((size_t)D * (H * VD) * 2);
    __bf16* w1T   = (__bf16*)alloc((size_t)FF * D * 2);
    __bf16* w2T   = (__bf16*)alloc((size_t)D * FF * 2);
    // activations
    __bf16* hs    = (__bf16*)alloc((size_t)S * D * 2);
    float*  q_aF  = (float*)alloc((size_t)S * QR * 4);
    __bf16* q_an  = (__bf16*)alloc((size_t)S * QR * 2);
    float*  qF    = (float*)alloc((size_t)S * H * QKD * 4);
    float*  kv_aF = (float*)alloc((size_t)S * (KVR + ROPE) * 4);
    __bf16* ckv_n = (__bf16*)alloc((size_t)S * KVR * 2);
    float*  kvF   = (float*)alloc((size_t)S * H * (128 + VD) * 4);
    __bf16* Qf    = (__bf16*)alloc((size_t)H * S * QKD * 2);
    __bf16* Kf    = (__bf16*)alloc((size_t)H * S * QKD * 2);
    __bf16* Vt    = (__bf16*)alloc((size_t)H * VD * S * 2);
    __bf16* att   = (__bf16*)alloc((size_t)S * (H * VD) * 2);
    float*  x2    = (float*)alloc((size_t)S * D * 4);
    __bf16* h2    = (__bf16*)alloc((size_t)S * D * 2);
    __bf16* a1    = (__bf16*)alloc((size_t)S * FF * 2);

    dim3 tb(32, 8);
    transpose_cvt_kernel<<<dim3(QR / 32, D / 32), tb, 0, stream>>>(q_a_w, q_aT, D, QR);
    transpose_cvt_kernel<<<dim3(H * QKD / 32, QR / 32), tb, 0, stream>>>(q_b_w, q_bT, QR, H * QKD);
    transpose_cvt_kernel<<<dim3((KVR + ROPE) / 32, D / 32), tb, 0, stream>>>(kv_a_w, kv_aT, D, KVR + ROPE);
    transpose_cvt_kernel<<<dim3(H * (128 + VD) / 32, KVR / 32), tb, 0, stream>>>(kv_b_w, kv_bT, KVR, H * (128 + VD));
    transpose_cvt_kernel<<<dim3(D / 32, H * VD / 32), tb, 0, stream>>>(o_w, oT, H * VD, D);
    transpose_cvt_kernel<<<dim3(FF / 32, D / 32), tb, 0, stream>>>(ffn_w1, w1T, D, FF);
    transpose_cvt_kernel<<<dim3(D / 32, FF / 32), tb, 0, stream>>>(ffn_w2, w2T, FF, D);

    // hs = rmsnorm(x, norm1_w)
    rmsnorm_bf16_kernel<<<S, 256, 0, stream>>>(x, norm1_w, hs, D, D, D);

    // q_a = hs @ q_a_w ; q_an = rmsnorm(q_a)
    gemm_bf16_kernel<0><<<dim3(QR / 128, S / 128), 256, 0, stream>>>(
        hs, q_aT, q_aF, nullptr, nullptr, nullptr, S, QR, D);
    rmsnorm_bf16_kernel<<<S, 256, 0, stream>>>(q_aF, q_a_ln_w, q_an, QR, QR, QR);

    // q = q_an @ q_b_w
    gemm_bf16_kernel<0><<<dim3(H * QKD / 128, S / 128), 256, 0, stream>>>(
        q_an, q_bT, qF, nullptr, nullptr, nullptr, S, H * QKD, QR);

    // kv_a = hs @ kv_a_w (N=576) ; ckv_n = rmsnorm(kv_a[:, :512])
    gemm_bf16_kernel<0><<<dim3((KVR + ROPE + 127) / 128, S / 128), 256, 0, stream>>>(
        hs, kv_aT, kv_aF, nullptr, nullptr, nullptr, S, KVR + ROPE, D);
    rmsnorm_bf16_kernel<<<S, 256, 0, stream>>>(kv_aF, kv_a_ln_w, ckv_n, KVR, KVR + ROPE, KVR);

    // kv = ckv_n @ kv_b_w
    gemm_bf16_kernel<0><<<dim3(H * (128 + VD) / 128, S / 128), 256, 0, stream>>>(
        ckv_n, kv_bT, kvF, nullptr, nullptr, nullptr, S, H * (128 + VD), KVR);

    // RoPE + head packing
    pack_q_kernel<<<dim3(S, H), 160, 0, stream>>>(qF, Qf);
    pack_k_kernel<<<dim3(S, H), 160, 0, stream>>>(kvF, kv_aF, Kf);
    pack_v_kernel<<<dim3(S, H), 128, 0, stream>>>(kvF, Vt);

    // causal flash attention
    attn_kernel<<<dim3(S / 64, H), 128, 0, stream>>>(Qf, Kf, Vt, att);

    // x2 = x + att @ o_w
    gemm_bf16_kernel<3><<<dim3(D / 128, S / 128), 256, 0, stream>>>(
        att, oT, x2, nullptr, nullptr, x, S, D, H * VD);

    // h2 = rmsnorm(x2, norm2_w)
    rmsnorm_bf16_kernel<<<S, 256, 0, stream>>>(x2, norm2_w, h2, D, D, D);

    // a1 = gelu(h2 @ w1 + b1)
    gemm_bf16_kernel<2><<<dim3(FF / 128, S / 128), 256, 0, stream>>>(
        h2, w1T, nullptr, a1, ffn_b1, nullptr, S, FF, D);

    // out = x2 + a1 @ w2 + b2
    gemm_bf16_kernel<4><<<dim3(D / 128, S / 128), 256, 0, stream>>>(
        a1, w2T, out, nullptr, ffn_b2, x2, S, D, FF);
}
